// Qwen35Model_1082331758940
// MI455X (gfx1250) — compile-verified, matrix operand order checked by reading, not run on hardware
//
#include <hip/hip_runtime.h>
#include <hip/hip_bf16.h>
#include <math.h>

// ---------------- model constants ----------------
#define T_SEQ  2048
#define DMODEL 1024
#define VOCAB  50257
#define HQn    8
#define Gn     2
#define HDn    256
#define RDn    64
#define HLn    16
#define DKn    128
#define DVn    128
#define Cch    6144      // 2*HL*DK + HL*DV
#define MMLP   4096
#define EPSF   1e-6f

// ---------------- types ----------------
typedef __attribute__((ext_vector_type(4)))  float          v4f;
typedef __attribute__((ext_vector_type(8)))  float          v8f;
typedef __attribute__((ext_vector_type(8)))  unsigned short v8us;
typedef __attribute__((ext_vector_type(16))) unsigned short v16us;
typedef __attribute__((ext_vector_type(16))) __bf16         v16bf;

// pack 8 floats (scaled by msk 0/1) into 8 bf16 — native convert, clang emits
// v_cvt_pk_bf16_f32 pairs on gfx1250 (RNE)
__device__ __forceinline__ v8us pack8(const float* v, float msk) {
  v8us o;
#pragma unroll
  for (int i = 0; i < 8; ++i) {
    __bf16 b = (__bf16)(v[i] * msk);
    o[i] = __builtin_bit_cast(unsigned short, b);
  }
  return o;
}
__device__ __forceinline__ v16bf load_frag(const unsigned short* p) {
  v8us lo = *reinterpret_cast<const v8us*>(p);
  v8us hi = *reinterpret_cast<const v8us*>(p + 16);
  v16us u = __builtin_shufflevector(lo, hi, 0,1,2,3,4,5,6,7,8,9,10,11,12,13,14,15);
  return __builtin_bit_cast(v16bf, u);
}
__device__ __forceinline__ float sigmoidf_(float x){ return 1.0f/(1.0f+__expf(-x)); }
__device__ __forceinline__ float siluf_(float x){ return x*sigmoidf_(x); }

// async DMA: one 32-bit element global -> LDS, tracked by ASYNCcnt
__device__ __forceinline__ void async_copy_f32(const float* gsrc, float* ldst) {
  unsigned l = (unsigned)(size_t)ldst;   // low 32 bits of LDS-aperture addr = LDS offset
  asm volatile("global_load_async_to_lds_b32 %0, %1, off"
               :: "v"(l), "v"(gsrc) : "memory");
}
__device__ __forceinline__ void wait_async0() {
  asm volatile("s_wait_asynccnt 0" ::: "memory");
}

// ---------------- WMMA GEMM: C[M,N] (+)= alpha * A[M,K] @ B ----------------
// A row-major (lda). B row-major [K,N] (ldb) if !TRANSB, else [N,K] (ldb).
// BM=BN=64, BK=64; 256 threads = 8 waves; each wave: 1 M-tile x 2 N-tiles,
// 4 v_wmma per K-tile. Requires K % 64 == 0 (true for all uses: 256..4096).
// Out-of-range rows/cols are handled branch-free: clamp address, zero by mask.
template<bool TRANSB, bool ACC>
__global__ __launch_bounds__(256)
void gemm_bf16(const float* __restrict__ A, int lda,
               const float* __restrict__ B, int ldb,
               float* __restrict__ C, int ldc,
               int M, int N, int K, float alpha)
{
  __shared__ unsigned short Al[64 * 64];   // [m][k]
  __shared__ unsigned short Bl[64 * 64];   // [n][k]
  const int tid  = threadIdx.x;
  const int bm   = blockIdx.y * 64;
  const int bn   = blockIdx.x * 64;
  const int wid  = tid >> 5, lane = tid & 31;
  const int mt   = wid & 3, np = wid >> 2;
  const int row16 = lane & 15, kg = lane >> 4;

  // staging indices (branch-free clamped addressing)
  const int ar  = tid >> 2,  ac  = (tid & 3) * 16;   // A: row, k base (16 wide)
  const int bcn = tid & 63,  brn = (tid >> 6) * 16;  // B NN: n, k base
  const int bct = tid >> 2,  brt = (tid & 3) * 16;   // B NT: n, k base

  const int   arow = (bm + ar  < M) ? (bm + ar)  : (M - 1);
  const float amsk = (bm + ar  < M) ? 1.0f : 0.0f;
  const int   bnnc = (bn + bcn < N) ? (bn + bcn) : (N - 1);
  const float bnnm = (bn + bcn < N) ? 1.0f : 0.0f;
  const int   bntc = (bn + bct < N) ? (bn + bct) : (N - 1);
  const float bntm = (bn + bct < N) ? 1.0f : 0.0f;

  v8f acc0 = {}, acc1 = {};

  for (int kb = 0; kb < K; kb += 64) {
    // ---------- stage A (64x64 fp32 -> bf16), unconditional vector loads ----------
    {
      const float* src = A + (size_t)arow * lda + kb + ac;
      float t[16];
      *(v4f*)(t)      = *(const v4f*)(src);
      *(v4f*)(t + 4)  = *(const v4f*)(src + 4);
      *(v4f*)(t + 8)  = *(const v4f*)(src + 8);
      *(v4f*)(t + 12) = *(const v4f*)(src + 12);
      *reinterpret_cast<v8us*>(Al + ar * 64 + ac)     = pack8(t, amsk);
      *reinterpret_cast<v8us*>(Al + ar * 64 + ac + 8) = pack8(t + 8, amsk);
      if (kb + 64 < K)
        __builtin_prefetch(src + 64, 0, 1);           // -> global_prefetch_b8
    }
    // ---------- stage B into Bl[n][k] ----------
    if (!TRANSB) {
      // B[K,N]: scalar reads (coalesced across lanes over n), vector LDS store
      const float* src = B + (size_t)(kb + brn) * ldb + bnnc;
      float t[16];
#pragma unroll
      for (int i = 0; i < 16; ++i) t[i] = src[(size_t)i * ldb];
      *reinterpret_cast<v8us*>(Bl + bcn * 64 + brn)     = pack8(t, bnnm);
      *reinterpret_cast<v8us*>(Bl + bcn * 64 + brn + 8) = pack8(t + 8, bnnm);
    } else {
      // B[N,K]: contiguous reads along k
      const float* src = B + (size_t)bntc * ldb + kb + brt;
      float t[16];
      *(v4f*)(t)      = *(const v4f*)(src);
      *(v4f*)(t + 4)  = *(const v4f*)(src + 4);
      *(v4f*)(t + 8)  = *(const v4f*)(src + 8);
      *(v4f*)(t + 12) = *(const v4f*)(src + 12);
      *reinterpret_cast<v8us*>(Bl + bct * 64 + brt)     = pack8(t, bntm);
      *reinterpret_cast<v8us*>(Bl + bct * 64 + brt + 8) = pack8(t + 8, bntm);
      if (kb + 64 < K)
        __builtin_prefetch(src + 64, 0, 1);
    }
    __syncthreads();

    // ---------- 2 K-substeps x 2 N-tiles = 4 WMMA ----------
#pragma unroll
    for (int kh = 0; kh < 2; ++kh) {
      const int ko = kh * 32 + kg * 8;
      v16bf a  = load_frag(Al + (mt * 16 + row16) * 64 + ko);
      v16bf b0 = load_frag(Bl + (np * 32 +      row16) * 64 + ko);
      v16bf b1 = load_frag(Bl + (np * 32 + 16 + row16) * 64 + ko);
      acc0 = __builtin_amdgcn_wmma_f32_16x16x32_bf16(false, a, false, b0, (short)0, acc0, false, false);
      acc1 = __builtin_amdgcn_wmma_f32_16x16x32_bf16(false, a, false, b1, (short)0, acc1, false, false);
    }
    __syncthreads();
  }

  // C/D layout: VGPR e -> row = e + 8*kg, col = row16
  const int rbase = bm + mt * 16 + kg * 8;
  const int c0 = bn + np * 32 + row16;
  const int c1 = c0 + 16;
#pragma unroll
  for (int e = 0; e < 8; ++e) {
    int r = rbase + e;
    if (r < M) {
      if (c0 < N) { size_t o = (size_t)r * ldc + c0; if (ACC) C[o] += alpha * acc0[e]; else C[o] = alpha * acc0[e]; }
      if (c1 < N) { size_t o = (size_t)r * ldc + c1; if (ACC) C[o] += alpha * acc1[e]; else C[o] = alpha * acc1[e]; }
    }
  }
}

// ---------------- elementwise / norm kernels ----------------
__global__ void embed_kernel(const int* __restrict__ inp, const float* __restrict__ emb,
                             float* __restrict__ x, int n) {
  int i = blockIdx.x * 256 + threadIdx.x;
  if (i >= n) return;
  int t = i / DMODEL, d = i % DMODEL;
  x[i] = emb[(size_t)inp[t] * DMODEL + d];
}

__global__ __launch_bounds__(256)
void rmsnorm_kernel(const float* __restrict__ in, const float* __restrict__ w,
                    float* __restrict__ out, int width) {
  __shared__ float red[256];
  size_t row = blockIdx.x;
  const float* xi = in + row * width;
  float* xo = out + row * width;
  int tid = threadIdx.x;
  float ss = 0.f;
  for (int d = tid; d < width; d += 256) { float v = xi[d]; ss += v * v; }
  red[tid] = ss; __syncthreads();
  for (int s = 128; s > 0; s >>= 1) { if (tid < s) red[tid] += red[tid + s]; __syncthreads(); }
  float sc = rsqrtf(red[0] / (float)width + EPSF);
  for (int d = tid; d < width; d += 256) xo[d] = xi[d] * sc * w[d];
}

__global__ void silu_mul_kernel(float* __restrict__ g, const float* __restrict__ u, int n) {
  int i = blockIdx.x * 256 + threadIdx.x;
  if (i < n) g[i] = siluf_(g[i]) * u[i];
}

// causal depthwise conv (CK=4 taps) + SiLU over [T, Cch]
__global__ void conv_silu_kernel(const float* __restrict__ qkv, const float* __restrict__ w,
                                 float* __restrict__ out, int n) {
  int i = blockIdx.x * 256 + threadIdx.x;
  if (i >= n) return;
  int t = i / Cch, c = i % Cch;
  float acc = 0.f;
#pragma unroll
  for (int j = 0; j < 4; ++j) {
    int ts = t + j - 3;
    if (ts >= 0) acc += qkv[(size_t)ts * Cch + c] * w[c * 4 + j];
  }
  out[i] = siluf_(acc);
}

// l2norm over width=128 on a strided slice of conv buffer (rows t*HL+h)
__global__ __launch_bounds__(128)
void l2norm_kernel(float* __restrict__ buf, int colOff, float scale) {
  __shared__ float red[128];
  int b = blockIdx.x;
  int t = b / HLn, h = b % HLn;
  float* p = buf + (size_t)t * Cch + colOff + h * DKn;
  int d = threadIdx.x;
  float v = p[d];
  red[d] = v * v; __syncthreads();
  for (int s = 64; s > 0; s >>= 1) { if (d < s) red[d] += red[d + s]; __syncthreads(); }
  p[d] = v * rsqrtf(red[0] + EPSF) * scale;
}

__global__ void beta_decay_kernel(const float* __restrict__ braw, const float* __restrict__ araw,
                                  const float* __restrict__ A_log, const float* __restrict__ dt_bias,
                                  float* __restrict__ beta, float* __restrict__ decay, int n) {
  int i = blockIdx.x * 256 + threadIdx.x;
  if (i >= n) return;
  int h = i & (HLn - 1);
  beta[i] = sigmoidf_(braw[i]);
  float x = araw[i] + dt_bias[h];
  float sp = (x > 20.f) ? x : log1pf(expf(x));
  decay[i] = expf(-expf(A_log[h]) * sp);
}

// delta-rule sequential scan: one workgroup per head, S[128][128] in LDS.
// q/k/v rows for step t+1 are DMA'd into a double buffer with
// global_load_async_to_lds_b32 (ASYNCcnt) while step t computes.
__global__ __launch_bounds__(128)
void delta_scan_kernel(const float* __restrict__ conv, const float* __restrict__ beta,
                       const float* __restrict__ decay, float* __restrict__ o) {
  extern __shared__ float sm[];
  float* S    = sm;                    // 128*128
  float* buf0 = sm + 128 * 128;        // [q|k|v] 3*128
  float* buf1 = buf0 + 3 * 128;        // [q|k|v] 3*128
  const int h = blockIdx.x, v = threadIdx.x;
  for (int k = 0; k < 128; ++k) S[k * 128 + v] = 0.f;
  // prime t = 0
  {
    const float* g = conv + h * 128 + v;
    async_copy_f32(g,        buf0 + v);
    async_copy_f32(g + 2048, buf0 + 128 + v);
    async_copy_f32(g + 4096, buf0 + 256 + v);
    wait_async0();
  }
  __syncthreads();
  for (int t = 0; t < T_SEQ; ++t) {
    float* cur = (t & 1) ? buf1 : buf0;
    float* nxt = (t & 1) ? buf0 : buf1;
    if (t + 1 < T_SEQ) {   // DMA next step's q/k/v while we compute
      const float* g = conv + (size_t)(t + 1) * Cch + h * 128 + v;
      async_copy_f32(g,        nxt + v);
      async_copy_f32(g + 2048, nxt + 128 + v);
      async_copy_f32(g + 4096, nxt + 256 + v);
    }
    const float* qs = cur;
    const float* ks = cur + 128;
    const float* vs = cur + 256;
    const float d = decay[t * HLn + h], b = beta[t * HLn + h];
    float vp = 0.f;
    for (int k = 0; k < 128; ++k) {
      float s = S[k * 128 + v] * d; S[k * 128 + v] = s;
      vp += ks[k] * s;
    }
    const float w = b * (vs[v] - vp);
    float ot = 0.f;
    for (int k = 0; k < 128; ++k) {
      float s = S[k * 128 + v] + ks[k] * w; S[k * 128 + v] = s;
      ot += qs[k] * s;
    }
    o[(size_t)t * 2048 + h * 128 + v] = ot;
    wait_async0();        // own wave's DMA done ...
    __syncthreads();      // ... and every wave's DMA done -> nxt valid, cur free
  }
}

// o = rmsnorm(o, out_norm) * silu(z), rows = T*HL, width 128, in place
__global__ __launch_bounds__(128)
void outnorm_gate_kernel(float* __restrict__ o, const float* __restrict__ w,
                         const float* __restrict__ z) {
  __shared__ float red[128];
  size_t row = blockIdx.x;
  float* p = o + row * 128;
  int d = threadIdx.x;
  float v = p[d];
  red[d] = v * v; __syncthreads();
  for (int s = 64; s > 0; s >>= 1) { if (d < s) red[d] += red[d + s]; __syncthreads(); }
  float sc = rsqrtf(red[0] / 128.0f + EPSF);
  p[d] = v * sc * w[d] * siluf_(z[row * 128 + d]);
}

// qg [T, HQ, 2*HD] -> qh [HQ,T,HD], gh [HQ,T,HD]
__global__ void split_qg_kernel(const float* __restrict__ qg, float* __restrict__ qh,
                                float* __restrict__ gh, int n) {
  int i = blockIdx.x * 256 + threadIdx.x;
  if (i >= n) return;
  int h = i / (T_SEQ * HDn), r = i % (T_SEQ * HDn);
  int t = r / HDn, d = r % HDn;
  size_t src = (size_t)t * (HQn * 2 * HDn) + h * (2 * HDn) + d;
  qh[i] = qg[src];
  gh[i] = qg[src + HDn];
}

// in [T, H, HD] -> out [H, T, HD]
__global__ void head_permute_kernel(const float* __restrict__ in, float* __restrict__ out,
                                    int H, int n) {
  int i = blockIdx.x * 256 + threadIdx.x;
  if (i >= n) return;
  int h = i / (T_SEQ * HDn), r = i % (T_SEQ * HDn);
  int t = r / HDn, d = r % HDn;
  out[i] = in[(size_t)t * (H * HDn) + h * HDn + d];
}

// per-(head,t) row: rmsnorm over HD=256 then RoPE on first 64 dims (in place)
__global__ __launch_bounds__(256)
void qknorm_rope_kernel(float* __restrict__ buf, const float* __restrict__ w) {
  __shared__ float red[256];
  __shared__ float arr[256];
  int b = blockIdx.x;
  int t = b % T_SEQ;
  float* p = buf + (size_t)b * HDn;
  int d = threadIdx.x;
  float x = p[d];
  red[d] = x * x; __syncthreads();
  for (int s = 128; s > 0; s >>= 1) { if (d < s) red[d] += red[d + s]; __syncthreads(); }
  float y = x * rsqrtf(red[0] / 256.0f + EPSF) * w[d];
  arr[d] = y; __syncthreads();
  float out = y;
  if (d < RDn) {
    int i = d & 31;
    float freq = __powf(1.0e7f, -(float)i / 32.0f);
    float ang = (float)t * freq;
    float rot = (d < 32) ? -arr[d + 32] : arr[d - 32];
    out = y * cosf(ang) + rot * sinf(ang);
  }
  __syncthreads();
  p[d] = out;
}

// causal softmax over row t of scores [T,T]; masked entries set to 0
__global__ __launch_bounds__(256)
void softmax_causal_kernel(float* __restrict__ sc) {
  __shared__ float red[256];
  int t = blockIdx.x, tid = threadIdx.x;
  float* row = sc + (size_t)t * T_SEQ;
  float m = -3.0e38f;
  for (int s = tid; s <= t; s += 256) m = fmaxf(m, row[s]);
  red[tid] = m; __syncthreads();
  for (int s = 128; s > 0; s >>= 1) { if (tid < s) red[tid] = fmaxf(red[tid], red[tid + s]); __syncthreads(); }
  m = red[0]; __syncthreads();
  float sum = 0.f;
  for (int s = tid; s <= t; s += 256) { float e = __expf(row[s] - m); row[s] = e; sum += e; }
  red[tid] = sum; __syncthreads();
  for (int s = 128; s > 0; s >>= 1) { if (tid < s) red[tid] += red[tid + s]; __syncthreads(); }
  float inv = 1.0f / red[0];
  for (int s = tid; s < T_SEQ; s += 256) row[s] = (s <= t) ? row[s] * inv : 0.0f;
}

// oh[HQ,T,HD] * sigmoid(gh) -> merged [T, HQ*HD]
__global__ void gate_merge_kernel(const float* __restrict__ oh, const float* __restrict__ gh,
                                  float* __restrict__ out, int n) {
  int i = blockIdx.x * 256 + threadIdx.x;
  if (i >= n) return;
  int h = i / (T_SEQ * HDn), r = i % (T_SEQ * HDn);
  int t = r / HDn, d = r % HDn;
  out[(size_t)t * (HQn * HDn) + h * HDn + d] = oh[i] * sigmoidf_(gh[i]);
}

// ---------------- host orchestration ----------------
static inline int ceilDiv(int a, int b) { return (a + b - 1) / b; }

static void launch_gemm(const float* A, int lda, const float* B, int ldb,
                        float* C, int ldc, int M, int N, int K,
                        float alpha, bool transB, bool acc, hipStream_t s) {
  dim3 grid(ceilDiv(N, 64), ceilDiv(M, 64)), blk(256);
  if (!transB && !acc) gemm_bf16<false, false><<<grid, blk, 0, s>>>(A, lda, B, ldb, C, ldc, M, N, K, alpha);
  else if (!transB &&  acc) gemm_bf16<false, true ><<<grid, blk, 0, s>>>(A, lda, B, ldb, C, ldc, M, N, K, alpha);
  else if ( transB && !acc) gemm_bf16<true,  false><<<grid, blk, 0, s>>>(A, lda, B, ldb, C, ldc, M, N, K, alpha);
  else                      gemm_bf16<true,  true ><<<grid, blk, 0, s>>>(A, lda, B, ldb, C, ldc, M, N, K, alpha);
}

struct DeltaP { const float *A_log,*a_w,*b_w,*conv_w,*down_w,*dt_bias,*gate_w,*k_w,
                            *norm1,*norm2,*o_w,*out_norm,*q_w,*up_w,*v_w,*z_w; };
struct AttnP  { const float *down_w,*gate_w,*k_norm,*k_w,*norm1,*norm2,*o_w,
                            *q_norm,*q_w,*up_w,*v_w; };

static void run_mlp(float* x, float* h, float* g1, float* u1,
                    const float* norm2, const float* gate_w, const float* up_w,
                    const float* down_w, hipStream_t s) {
  rmsnorm_kernel<<<T_SEQ, 256, 0, s>>>(x, norm2, h, DMODEL);
  launch_gemm(h, DMODEL, gate_w, MMLP, g1, MMLP, T_SEQ, MMLP, DMODEL, 1.f, false, false, s);
  launch_gemm(h, DMODEL, up_w,   MMLP, u1, MMLP, T_SEQ, MMLP, DMODEL, 1.f, false, false, s);
  int n = T_SEQ * MMLP;
  silu_mul_kernel<<<ceilDiv(n, 256), 256, 0, s>>>(g1, u1, n);
  launch_gemm(g1, MMLP, down_w, DMODEL, x, DMODEL, T_SEQ, DMODEL, MMLP, 1.f, false, true, s);
}

extern "C" void kernel_launch(void* const* d_in, const int* in_sizes, int n_in,
                              void* d_out, int out_size, void* d_ws, size_t ws_size,
                              hipStream_t stream) {
  (void)in_sizes; (void)n_in; (void)out_size; (void)ws_size;
  static const int LTYPE[8] = {0, 0, 0, 1, 0, 0, 0, 1};

  // ---- unpack flattened params (jax pytree: dict keys in sorted order) ----
  int cur = 0;
  const int* inp = (const int*)d_in[cur++];
  DeltaP dp[6]; AttnP ap[2];
  int kindIdx[8]; int di = 0, ai = 0;
  for (int b = 0; b < 8; ++b) {
    if (LTYPE[b] == 0) {
      DeltaP& p = dp[di];
      p.A_log   = (const float*)d_in[cur++];
      p.a_w     = (const float*)d_in[cur++];
      p.b_w     = (const float*)d_in[cur++];
      p.conv_w  = (const float*)d_in[cur++];
      p.down_w  = (const float*)d_in[cur++];
      p.dt_bias = (const float*)d_in[cur++];
      p.gate_w  = (const float*)d_in[cur++];
      p.k_w     = (const float*)d_in[cur++];
      p.norm1   = (const float*)d_in[cur++];
      p.norm2   = (const float*)d_in[cur++];
      p.o_w     = (const float*)d_in[cur++];
      p.out_norm= (const float*)d_in[cur++];
      p.q_w     = (const float*)d_in[cur++];
      p.up_w    = (const float*)d_in[cur++];
      p.v_w     = (const float*)d_in[cur++];
      p.z_w     = (const float*)d_in[cur++];
      kindIdx[b] = di++;
    } else {
      AttnP& p = ap[ai];
      p.down_w  = (const float*)d_in[cur++];
      p.gate_w  = (const float*)d_in[cur++];
      p.k_norm  = (const float*)d_in[cur++];
      p.k_w     = (const float*)d_in[cur++];
      p.norm1   = (const float*)d_in[cur++];
      p.norm2   = (const float*)d_in[cur++];
      p.o_w     = (const float*)d_in[cur++];
      p.q_norm  = (const float*)d_in[cur++];
      p.q_w     = (const float*)d_in[cur++];
      p.up_w    = (const float*)d_in[cur++];
      p.v_w     = (const float*)d_in[cur++];
      kindIdx[b] = ai++;
    }
  }
  const float* emb        = (const float*)d_in[cur++];
  const float* final_norm = (const float*)d_in[cur++];
  const float* final_proj = (const float*)d_in[cur++];

  // ---- workspace layout (floats) ----
  float* base = (float*)d_ws;
  const size_t TD = (size_t)T_SEQ * DMODEL;
  float* x  = base;
  float* h  = base + TD;
  float* ar = base + 2 * TD;                // phase-shared arena
  // delta phase
  float* qkv  = ar;
  float* conv = qkv  + (size_t)T_SEQ * Cch;
  float* zbuf = conv + (size_t)T_SEQ * Cch;
  float* obuf = zbuf + (size_t)T_SEQ * 2048;
  float* braw = obuf + (size_t)T_SEQ * 2048;
  float* araw = braw + (size_t)T_SEQ * HLn;
  float* beta = araw + (size_t)T_SEQ * HLn;
  float* dec  = beta + (size_t)T_SEQ * HLn;
  // attn phase (aliases delta phase)
  float* qg = ar;
  float* qh = qg + (size_t)T_SEQ * 4096;
  float* gh = qh + (size_t)HQn * T_SEQ * HDn;
  float* kh = gh + (size_t)HQn * T_SEQ * HDn;
  float* vh = kh + (size_t)Gn * T_SEQ * HDn;
  float* sc = vh + (size_t)Gn * T_SEQ * HDn;
  float* oh = sc + (size_t)T_SEQ * T_SEQ;
  float* mg = oh + (size_t)HQn * T_SEQ * HDn;
  float* kt = sc;                            // temp proj [T,512], reused before sc
  float* vt = sc + (size_t)T_SEQ * 512;
  // mlp phase (aliases arena)
  float* g1 = ar;
  float* u1 = ar + (size_t)T_SEQ * MMLP;

  // ---- embedding ----
  {
    int n = (int)TD;
    embed_kernel<<<ceilDiv(n, 256), 256, 0, stream>>>(inp, emb, x, n);
  }

  // ---- layers ----
  for (int b = 0; b < 8; ++b) {
    if (LTYPE[b] == 0) {
      const DeltaP& p = dp[kindIdx[b]];
      rmsnorm_kernel<<<T_SEQ, 256, 0, stream>>>(x, p.norm1, h, DMODEL);
      // projections into concatenated [q|k|v] buffer (ldc = Cch)
      launch_gemm(h, DMODEL, p.q_w, 2048, qkv,        Cch, T_SEQ, 2048, DMODEL, 1.f, false, false, stream);
      launch_gemm(h, DMODEL, p.k_w, 2048, qkv + 2048, Cch, T_SEQ, 2048, DMODEL, 1.f, false, false, stream);
      launch_gemm(h, DMODEL, p.v_w, 2048, qkv + 4096, Cch, T_SEQ, 2048, DMODEL, 1.f, false, false, stream);
      launch_gemm(h, DMODEL, p.z_w, 2048, zbuf, 2048, T_SEQ, 2048, DMODEL, 1.f, false, false, stream);
      launch_gemm(h, DMODEL, p.b_w, HLn,  braw, HLn,  T_SEQ, HLn,  DMODEL, 1.f, false, false, stream);
      launch_gemm(h, DMODEL, p.a_w, HLn,  araw, HLn,  T_SEQ, HLn,  DMODEL, 1.f, false, false, stream);
      // conv + silu
      {
        int n = T_SEQ * Cch;
        conv_silu_kernel<<<ceilDiv(n, 256), 256, 0, stream>>>(qkv, p.conv_w, conv, n);
      }
      // l2norm q (scaled by DK^-0.5) and k
      l2norm_kernel<<<T_SEQ * HLn, 128, 0, stream>>>(conv, 0,    0.08838834764831845f);
      l2norm_kernel<<<T_SEQ * HLn, 128, 0, stream>>>(conv, 2048, 1.0f);
      // beta / decay
      {
        int n = T_SEQ * HLn;
        beta_decay_kernel<<<ceilDiv(n, 256), 256, 0, stream>>>(braw, araw, p.A_log, p.dt_bias, beta, dec, n);
      }
      // sequential delta scan, S + double-buffered async q/k/v in LDS
      delta_scan_kernel<<<HLn, 128, (128 * 128 + 6 * 128) * sizeof(float), stream>>>(conv, beta, dec, obuf);
      // out-norm * silu(z)
      outnorm_gate_kernel<<<T_SEQ * HLn, 128, 0, stream>>>(obuf, p.out_norm, zbuf);
      // x += o @ o_w
      launch_gemm(obuf, 2048, p.o_w, DMODEL, x, DMODEL, T_SEQ, DMODEL, 2048, 1.f, false, true, stream);
      run_mlp(x, h, g1, u1, p.norm2, p.gate_w, p.up_w, p.down_w, stream);
    } else {
      const AttnP& p = ap[kindIdx[b]];
      rmsnorm_kernel<<<T_SEQ, 256, 0, stream>>>(x, p.norm1, h, DMODEL);
      launch_gemm(h, DMODEL, p.q_w, 4096, qg, 4096, T_SEQ, 4096, DMODEL, 1.f, false, false, stream);
      launch_gemm(h, DMODEL, p.k_w, 512,  kt, 512,  T_SEQ, 512,  DMODEL, 1.f, false, false, stream);
      launch_gemm(h, DMODEL, p.v_w, 512,  vt, 512,  T_SEQ, 512,  DMODEL, 1.f, false, false, stream);
      {
        int nq = HQn * T_SEQ * HDn;
        split_qg_kernel<<<ceilDiv(nq, 256), 256, 0, stream>>>(qg, qh, gh, nq);
        int nk = Gn * T_SEQ * HDn;
        head_permute_kernel<<<ceilDiv(nk, 256), 256, 0, stream>>>(kt, kh, Gn, nk);
        head_permute_kernel<<<ceilDiv(nk, 256), 256, 0, stream>>>(vt, vh, Gn, nk);
      }
      qknorm_rope_kernel<<<HQn * T_SEQ, 256, 0, stream>>>(qh, p.q_norm);
      qknorm_rope_kernel<<<Gn  * T_SEQ, 256, 0, stream>>>(kh, p.k_norm);
      // per-head attention (stream-serialized; scores buffer reused)
      for (int hq = 0; hq < HQn; ++hq) {
        int g = hq / (HQn / Gn);
        const float* qp = qh + (size_t)hq * T_SEQ * HDn;
        const float* kp = kh + (size_t)g  * T_SEQ * HDn;
        const float* vp = vh + (size_t)g  * T_SEQ * HDn;
        float* op = oh + (size_t)hq * T_SEQ * HDn;
        // scores = q @ k^T * HD^-0.5   (B = k [T,HD] -> transB)
        launch_gemm(qp, HDn, kp, HDn, sc, T_SEQ, T_SEQ, T_SEQ, HDn, 0.0625f, true, false, stream);
        softmax_causal_kernel<<<T_SEQ, 256, 0, stream>>>(sc);
        // o = P @ v   (B = v [T(=K), HD(=N)] row-major -> NN)
        launch_gemm(sc, T_SEQ, vp, HDn, op, HDn, T_SEQ, HDn, T_SEQ, 1.f, false, false, stream);
      }
      {
        int n = HQn * T_SEQ * HDn;
        gate_merge_kernel<<<ceilDiv(n, 256), 256, 0, stream>>>(oh, gh, mg, n);
      }
      launch_gemm(mg, 2048, p.o_w, DMODEL, x, DMODEL, T_SEQ, DMODEL, 2048, 1.f, false, true, stream);
      run_mlp(x, h, g1, u1, p.norm2, p.gate_w, p.up_w, p.down_w, stream);
    }
  }

  // ---- final norm + vocab projection ----
  rmsnorm_kernel<<<T_SEQ, 256, 0, stream>>>(x, final_norm, h, DMODEL);
  launch_gemm(h, DMODEL, final_proj, VOCAB, (float*)d_out, VOCAB,
              T_SEQ, VOCAB, DMODEL, 1.f, false, false, stream);
}
